// Network_52613349376276
// MI455X (gfx1250) — matrix operand order repaired and checked by reading, not compiled, not run on hardware
//
#include <hip/hip_runtime.h>

// ---------------------------------------------------------------------------
// 4-layer LSTM for MI455X (gfx1250, wave32, WMMA).
//
//   * Input projection x_t @ W^T + b batched over all timesteps as a parallel
//     f16 WMMA GEMM ([T*B,256]x[256,1024]); result stored PRE-SWIZZLED in
//     C/D-fragment order so the sequential scan re-loads each accumulator
//     tile with one coalesced global_load_b128 per lane.
//   * Recurrence: one persistent 1024-thread workgroup (32 wave32s, 1 WGP)
//     per layer. h (f16, bank-padded), c (f32), and the 32x1024 f32 gate
//     pre-activations live in 176KB LDS (CDNA5: 320KB/WGP). Each wave owns
//     4 of 128 16x16 tiles -> 32 v_wmma_f32_16x16x32_f16 per step; R streams
//     from the 192MB L2 (4MB of weights = permanently resident).
//   * c-state staging uses GLOBAL_LOAD_ASYNC_TO_LDS_B128 + s_wait_asynccnt.
// ---------------------------------------------------------------------------

typedef __attribute__((ext_vector_type(16))) _Float16 v16h;
typedef __attribute__((ext_vector_type(8)))  _Float16 v8h;
typedef __attribute__((ext_vector_type(8)))  float    v8f;

#define HID    256
#define BATCH  32
#define SEQ    1024
#define LAYERS 4
#define GATES  1024          // 4*HID
#define HPAD   264           // h row pad: 264 halves = 132 dwords -> +4 banks/row

__device__ __forceinline__ float sigmoid_(float x) { return 1.0f / (1.0f + __expf(-x)); }
__device__ __forceinline__ float tanh_(float x)    { return 1.0f - 2.0f / (__expf(2.0f * x) + 1.0f); }

// A fragment (16x32 f16). rowbase -> element [m][kc*32], row-major.
// ISA: element e holds k = (e&8)*2 + hi*8 + (e&7).
__device__ __forceinline__ v16h load_a_frag(const _Float16* rowbase, int hi) {
    const _Float16* p = rowbase + hi * 8;
    v8h p0 = *(const v8h*)(p);
    v8h p1 = *(const v8h*)(p + 16);
    v16h a;
#pragma unroll
    for (int e = 0; e < 8; ++e) { a[e] = p0[e]; a[8 + e] = p1[e]; }
    return a;
}

// B fragment (32x16 f16), B[k][n] = M[n][k], M row-major [N][K].
// rowbase -> element [n][kc*32]. ISA: element e holds k = (e&8)*2 + hi*8 + (e&7).
__device__ __forceinline__ v16h load_b_frag(const _Float16* rowbase, int hi) {
    const _Float16* p = rowbase + hi * 8;
    v8h q0 = *(const v8h*)(p);
    v8h q1 = *(const v8h*)(p + 16);
    v16h b;
#pragma unroll
    for (int e = 0; e < 8; ++e) { b[e] = q0[e]; b[8 + e] = q1[e]; }
    return b;
}

// Async global->LDS copy of 16B per lane (GV mode), tracked by ASYNCcnt.
__device__ __forceinline__ void async_copy_b128(unsigned lds_addr, const void* gptr) {
    unsigned long long g = (unsigned long long)(uintptr_t)gptr;
    asm volatile("global_load_async_to_lds_b128 %0, %1, off"
                 :: "v"(lds_addr), "v"(g) : "memory");
}
__device__ __forceinline__ void wait_asynccnt0() {
    asm volatile("s_wait_asynccnt 0" ::: "memory");
}

// ---------------------------------------------------------------------------
// fp32 -> f16 conversion (weights + layer-0 input)
// ---------------------------------------------------------------------------
__global__ void cvt_f32_f16(const float* __restrict__ src, _Float16* __restrict__ dst, int n) {
    for (int i = blockIdx.x * blockDim.x + threadIdx.x; i < n; i += gridDim.x * blockDim.x)
        dst[i] = (_Float16)src[i];
}

// ---------------------------------------------------------------------------
// xw = inp @ W_l^T + bias_l, output stored per 16x16 tile in C/D fragment
// order: xws[(rowTile*64 + ct)*256 + lane*8 + v]  (f16).
// Grid: x = 64 col tiles, y = 64; block = 256 (8 waves); wave = 4 row tiles.
// ---------------------------------------------------------------------------
__global__ __launch_bounds__(256) void input_gemm(const _Float16* __restrict__ X,
                                                  const _Float16* __restrict__ Wh,
                                                  const float* __restrict__ bias,
                                                  _Float16* __restrict__ xws) {
    const int tid = threadIdx.x;
    const int ln  = tid & 31, wv = tid >> 5;
    const int lo  = ln & 15,  hi = ln >> 4;
    const int ct  = blockIdx.x;                       // column tile 0..63
    const int rt0 = (blockIdx.y * 8 + wv) * 4;        // first of 4 row tiles
    const int nG  = ct * 16 + lo;                     // this lane's gate column

    const float bv = bias[nG];
    v8f acc[4];
#pragma unroll
    for (int r = 0; r < 4; ++r)
#pragma unroll
        for (int v = 0; v < 8; ++v) acc[r][v] = bv;

#pragma unroll
    for (int kc = 0; kc < 8; ++kc) {
        v16h b = load_b_frag(Wh + (size_t)nG * HID + kc * 32, hi);
#pragma unroll
        for (int r = 0; r < 4; ++r) {
            v16h a = load_a_frag(X + (size_t)((rt0 + r) * 16 + lo) * HID + kc * 32, hi);
            acc[r] = __builtin_amdgcn_wmma_f32_16x16x32_f16(false, a, false, b,
                                                            (short)0, acc[r], false, false);
        }
    }
#pragma unroll
    for (int r = 0; r < 4; ++r) {
        v8h st;
#pragma unroll
        for (int v = 0; v < 8; ++v) st[v] = (_Float16)acc[r][v];
        *(v8h*)(xws + ((size_t)(rt0 + r) * 64 + ct) * 256 + ln * 8) = st;
    }
}

// ---------------------------------------------------------------------------
// Persistent recurrent scan for one layer: single block, 1024 threads.
// Wave w owns gate-column tiles {2w, 2w+1} x batch-row tiles {0,1}.
// ---------------------------------------------------------------------------
__global__ __launch_bounds__(1024) void lstm_scan(const _Float16* __restrict__ Rh,   // [GATES][HID] f16
                                                  const _Float16* __restrict__ xws,  // swizzled gate tiles
                                                  const float* __restrict__ h0,      // [BATCH*HID]
                                                  const float* __restrict__ c0,      // [BATCH*HID]
                                                  _Float16* __restrict__ out16,      // null on top layer
                                                  float* __restrict__ out32)         // null except top layer
{
    extern __shared__ char smem[];
    _Float16* h_sh = (_Float16*)smem;                                    // [BATCH][HPAD]  16.9KB
    float*    c_sh = (float*)(smem + BATCH * HPAD * 2);                  // [BATCH][HID]   32KB
    float*    g_sh = (float*)(smem + BATCH * HPAD * 2 + BATCH * HID * 4);// [BATCH][GATES] 128KB

    const int tid = threadIdx.x;
    const int ln  = tid & 31, wv = tid >> 5;
    const int lo  = ln & 15,  hi = ln >> 4;
    const int ct0 = 2 * wv;                          // first owned column tile

    // ---- stage c0 into LDS with async global->LDS DMA (32KB, 2x16B/thread) --
    {
        const unsigned c_lds = (unsigned)(uintptr_t)(void*)c_sh;
#pragma unroll
        for (int q = 0; q < 2; ++q) {
            const unsigned byteoff = (unsigned)(q * 1024 + tid) * 16u;
            async_copy_b128(c_lds + byteoff, (const char*)c0 + byteoff);
        }
    }
    // h needs f32->f16 conversion: plain loads.
    for (int e = tid; e < BATCH * HID; e += 1024) {
        const int b = e >> 8, j = e & 255;
        h_sh[b * HPAD + j] = (_Float16)h0[e];
    }
    wait_asynccnt0();
    __syncthreads();

    for (int t = 0; t < SEQ; ++t) {
        // ---- init accumulators: one b128 fragment load per tile ----
        v8f acc[2][2];
#pragma unroll
        for (int mt = 0; mt < 2; ++mt)
#pragma unroll
            for (int t2 = 0; t2 < 2; ++t2) {
                const v8h q = *(const v8h*)(xws + ((size_t)(t * 2 + mt) * 64 + ct0 + t2) * 256 + ln * 8);
#pragma unroll
                for (int v = 0; v < 8; ++v) acc[mt][t2][v] = (float)q[v];
            }

        if (t + 1 < SEQ) {  // uniform; warm next step's fragments into cache
#pragma unroll
            for (int mt = 0; mt < 2; ++mt)
                __builtin_prefetch(xws + ((size_t)((t + 1) * 2 + mt) * 64 + ct0) * 256 + ln * 8, 0, 3);
        }

        // ---- gates += h @ R^T : 8 K-chunks x 4 tiles = 32 WMMA / wave ----
#pragma unroll
        for (int kc = 0; kc < 8; ++kc) {
            v16h a[2];
#pragma unroll
            for (int mt = 0; mt < 2; ++mt)
                a[mt] = load_a_frag(h_sh + (size_t)(mt * 16 + lo) * HPAD + kc * 32, hi);
#pragma unroll
            for (int t2 = 0; t2 < 2; ++t2) {
                v16h b = load_b_frag(Rh + (size_t)((ct0 + t2) * 16 + lo) * HID + kc * 32, hi);
#pragma unroll
                for (int mt = 0; mt < 2; ++mt)
                    acc[mt][t2] = __builtin_amdgcn_wmma_f32_16x16x32_f16(false, a[mt], false, b,
                                                                         (short)0, acc[mt][t2],
                                                                         false, false);
            }
        }

        // ---- publish gate pre-activations to LDS ----
#pragma unroll
        for (int mt = 0; mt < 2; ++mt)
#pragma unroll
            for (int t2 = 0; t2 < 2; ++t2)
#pragma unroll
                for (int v = 0; v < 8; ++v)
                    g_sh[(mt * 16 + v + 8 * hi) * GATES + (ct0 + t2) * 16 + lo] = acc[mt][t2][v];
        __syncthreads();

        // ---- pointwise LSTM cell: 8192 elements / 1024 threads ----
        for (int e = tid; e < BATCH * HID; e += 1024) {
            const int b = e >> 8, j = e & 255;
            const float* gr = g_sh + b * GATES;
            const float ig = sigmoid_(gr[j]);
            const float fg = sigmoid_(gr[256 + j]);
            const float gg = tanh_(gr[512 + j]);
            const float og = sigmoid_(gr[768 + j]);
            const float cn = fg * c_sh[e] + ig * gg;
            const float hn = og * tanh_(cn);
            c_sh[e] = cn;
            h_sh[b * HPAD + j] = (_Float16)hn;
            if (out16) out16[(size_t)t * BATCH * HID + e] = (_Float16)hn;
            if (out32) out32[(size_t)t * BATCH * HID + e] = hn;
        }
        __syncthreads();
    }
}

// ---------------------------------------------------------------------------
// Host launcher.
// Workspace layout (100MB total):
//   Wh (2MB) | Rh (2MB) | inpA (16MB) | inpB (16MB) | xws (64MB)
// ---------------------------------------------------------------------------
extern "C" void kernel_launch(void* const* d_in, const int* in_sizes, int n_in,
                              void* d_out, int out_size, void* d_ws, size_t ws_size,
                              hipStream_t stream) {
    (void)in_sizes; (void)n_in; (void)out_size; (void)ws_size;
    const float* h_data = (const float*)d_in[0];
    const float* x_data = (const float*)d_in[1];
    const float* c_data = (const float*)d_in[2];
    const float* W      = (const float*)d_in[3];
    const float* R      = (const float*)d_in[4];
    const float* bias   = (const float*)d_in[5];
    float* out = (float*)d_out;

    char* ws = (char*)d_ws;
    size_t off = 0;
    _Float16* Wh   = (_Float16*)(ws + off); off += (size_t)LAYERS * GATES * HID * sizeof(_Float16);
    _Float16* Rh   = (_Float16*)(ws + off); off += (size_t)LAYERS * GATES * HID * sizeof(_Float16);
    _Float16* inpA = (_Float16*)(ws + off); off += (size_t)SEQ * BATCH * HID * sizeof(_Float16);
    _Float16* inpB = (_Float16*)(ws + off); off += (size_t)SEQ * BATCH * HID * sizeof(_Float16);
    _Float16* xws  = (_Float16*)(ws + off); off += (size_t)SEQ * BATCH * GATES * sizeof(_Float16);

    cvt_f32_f16<<<512, 256, 0, stream>>>(W, Wh, LAYERS * GATES * HID);
    cvt_f32_f16<<<512, 256, 0, stream>>>(R, Rh, LAYERS * GATES * HID);
    cvt_f32_f16<<<2048, 256, 0, stream>>>(x_data, inpA, SEQ * BATCH * HID);

    const size_t lds_bytes = (size_t)BATCH * HPAD * 2 + (size_t)BATCH * HID * 4
                           + (size_t)BATCH * GATES * 4;   // 176.5KB < 320KB/WGP
    (void)hipFuncSetAttribute(reinterpret_cast<const void*>(lstm_scan),
                              hipFuncAttributeMaxDynamicSharedMemorySize, (int)lds_bytes);

    _Float16* cur = inpA;
    _Float16* nxt = inpB;
    for (int l = 0; l < LAYERS; ++l) {
        const bool top = (l == LAYERS - 1);
        input_gemm<<<dim3(64, 64), 256, 0, stream>>>(cur,
                                                     Wh + (size_t)l * GATES * HID,
                                                     bias + (size_t)l * GATES, xws);
        lstm_scan<<<1, 1024, lds_bytes, stream>>>(Rh + (size_t)l * GATES * HID, xws,
                                                  h_data + (size_t)l * (SEQ + 1) * BATCH * HID,
                                                  c_data + (size_t)l * (SEQ + 1) * BATCH * HID,
                                                  top ? nullptr : nxt,
                                                  top ? out : nullptr);
        _Float16* tmp = cur; cur = nxt; nxt = tmp;
    }
}
